// RPL_Asymm_3d_spconv_33500744909242
// MI455X (gfx1250) — compile-verified, matrix operand order checked by reading, not run on hardware
//
#include <hip/hip_runtime.h>

#define NPTS 131072
#define CCH  256
#define EPSV 1e-5f
#define LDSTRIDE 264   // 256 halves + 8 pad halves per staged row

typedef __attribute__((ext_vector_type(16))) _Float16 v16h;
typedef __attribute__((ext_vector_type(8)))  float    v8f;

union V16H { v16h v; uint4 q[2]; };

#define WMMA_F16(A, B, C) \
  __builtin_amdgcn_wmma_f32_16x16x32_f16(false, (A), false, (B), (short)0, (C), false, false)

// ---------------------------------------------------------------------------
// Pack W[3][3][256][256] f32 -> f16 B-fragments in WMMA register order.
// Fragment layout (16-bit B 32x16, ISA 7.12.2): lane<16 -> col=nt*16+lane,
// halves 0..7 = K 0..7 (+8 for lanes>=16), halves 8..15 = K 16..23 (+8).
// Flat dst index = ((((axis*3+tap)*8+kt)*16+nt)*512) + lane*16 + h
// ---------------------------------------------------------------------------
__global__ void wpack_kernel(const float* __restrict__ W1,
                             const float* __restrict__ W2,
                             const float* __restrict__ W3,
                             _Float16* __restrict__ wpack) {
  int e = blockIdx.x * blockDim.x + threadIdx.x;   // [0, 3*196608)
  int a = e / 196608; int r = e % 196608;
  int t  = r / 65536;  r %= 65536;
  int kt = r / 8192;   r %= 8192;
  int nt = r / 512;    int le = r % 512;
  int lane = le >> 4, h = le & 15;
  int v = h >> 1, p = h & 1;
  int kk = kt * 32 + ((lane >= 16) ? 8 : 0)
         + ((v < 4) ? (v * 2 + p) : (16 + (v - 4) * 2 + p));
  int col = nt * 16 + (lane & 15);
  const float* W = (a == 0) ? W1 : ((a == 1) ? W2 : W3);
  wpack[e] = (_Float16)W[t * 65536 + kk * 256 + col];
}

// ---------------------------------------------------------------------------
// Gather-GEMM for one axis: h[rows, 256] = sum_tap gather(F)*W[tap]
// Block: 256 threads (8 waves), 64 rows x 256 cols.
// Wave w: M-tiles {2*(w>>2), 2*(w>>2)+1}, N-tiles 4*(w&3)..+3 -> 8 accs.
// B fragments double-buffered (load kt+1 before WMMAs of kt) so no WMMA
// waits on its own loads; each B fragment feeds 2 WMMAs (M reuse).
// Also accumulates per-channel sum / sumsq into stats[0..255] / [256..511].
// ---------------------------------------------------------------------------
__global__ void __launch_bounds__(256) spconv_gemm(
    const float* __restrict__ feat,      // [N,256] f32
    const _Float16* __restrict__ wp,     // this axis: [3][8][16][512] f16
    const int* __restrict__ nbr,         // [N,3] int32, >=N -> missing
    float* __restrict__ hout,            // [N,256] f32
    float* __restrict__ stats) {         // [2,256] f32 (pre-zeroed)
  __shared__ _Float16 smem[64 * LDSTRIDE];

  const int tid  = threadIdx.x;
  const int lane = tid & 31;
  const int w    = tid >> 5;
  const int mt0  = (w >> 2) << 1;      // 0 or 2 (two 16-row tiles each)
  const int nt0  = (w & 3) << 2;       // 0,4,8,12
  const int row0 = blockIdx.x * 64;

  v8f acc[8];
  #pragma unroll
  for (int i = 0; i < 8; ++i) acc[i] = (v8f){};

  const int srow = tid >> 2;   // staged row 0..63
  const int sseg = tid & 3;    // 64-column segment

  const int arow  = mt0 * 16 + (lane & 15);
  const int khalf = (lane >> 4) << 3;

  for (int tap = 0; tap < 3; ++tap) {
    __syncthreads();
    // ---- stage 64 gathered rows into LDS as f16 ----
    {
      int gi = nbr[(row0 + srow) * 3 + tap];
      __align__(16) _Float16 tmp[64];
      if (gi >= 0 && gi < NPTS) {
        const float4* src = (const float4*)(feat + (size_t)gi * CCH + sseg * 64);
        #pragma unroll
        for (int i = 0; i < 16; ++i) {
          float4 v4 = src[i];
          tmp[i * 4 + 0] = (_Float16)v4.x;
          tmp[i * 4 + 1] = (_Float16)v4.y;
          tmp[i * 4 + 2] = (_Float16)v4.z;
          tmp[i * 4 + 3] = (_Float16)v4.w;
        }
      } else {
        #pragma unroll
        for (int i = 0; i < 64; ++i) tmp[i] = (_Float16)0.f;
      }
      _Float16* dst = &smem[srow * LDSTRIDE + sseg * 64];
      #pragma unroll
      for (int i = 0; i < 8; ++i)
        ((uint4*)dst)[i] = ((const uint4*)tmp)[i];
    }
    __syncthreads();

    const _Float16* bbase = wp + (size_t)tap * (8 * 16 * 512) + (size_t)lane * 16;

    // ---- prologue: B fragments for kt = 0 ----
    v16h b0 = *(const v16h*)(bbase + (size_t)(nt0 + 0) * 512);
    v16h b1 = *(const v16h*)(bbase + (size_t)(nt0 + 1) * 512);
    v16h b2 = *(const v16h*)(bbase + (size_t)(nt0 + 2) * 512);
    v16h b3 = *(const v16h*)(bbase + (size_t)(nt0 + 3) * 512);

    #pragma unroll
    for (int kt = 0; kt < 8; ++kt) {
      // A fragments for both M-tiles (LDS)
      V16H A0, A1;
      const _Float16* ap = &smem[arow * LDSTRIDE + kt * 32 + khalf];
      A0.q[0] = *(const uint4*)ap;
      A0.q[1] = *(const uint4*)(ap + 16);
      const _Float16* ap2 = ap + 16 * LDSTRIDE;
      A1.q[0] = *(const uint4*)ap2;
      A1.q[1] = *(const uint4*)(ap2 + 16);

      // prefetch next k-step's B fragments into fresh registers
      v16h n0, n1, n2, n3;
      if (kt < 7) {
        const _Float16* bn = bbase + (size_t)(kt + 1) * (16 * 512);
        n0 = *(const v16h*)(bn + (size_t)(nt0 + 0) * 512);
        n1 = *(const v16h*)(bn + (size_t)(nt0 + 1) * 512);
        n2 = *(const v16h*)(bn + (size_t)(nt0 + 2) * 512);
        n3 = *(const v16h*)(bn + (size_t)(nt0 + 3) * 512);
      } else {
        n0 = b0; n1 = b1; n2 = b2; n3 = b3;
      }

      acc[0] = WMMA_F16(A0.v, b0, acc[0]);
      acc[4] = WMMA_F16(A1.v, b0, acc[4]);
      acc[1] = WMMA_F16(A0.v, b1, acc[1]);
      acc[5] = WMMA_F16(A1.v, b1, acc[5]);
      acc[2] = WMMA_F16(A0.v, b2, acc[2]);
      acc[6] = WMMA_F16(A1.v, b2, acc[6]);
      acc[3] = WMMA_F16(A0.v, b3, acc[3]);
      acc[7] = WMMA_F16(A1.v, b3, acc[7]);

      b0 = n0; b1 = n1; b2 = n2; b3 = n3;
    }
  }

  // ---- store h + fused BN statistics ----
  // Acc layout: VGPR r holds M=r (lanes 0-15) / M=r+8 (lanes 16-31).
  #pragma unroll
  for (int m = 0; m < 2; ++m) {
    const int rbase = row0 + (mt0 + m) * 16 + ((lane >> 4) << 3);
    #pragma unroll
    for (int j = 0; j < 4; ++j) {
      int c = (nt0 + j) * 16 + (lane & 15);
      v8f a = acc[m * 4 + j];
      float s = 0.f, q = 0.f;
      #pragma unroll
      for (int r = 0; r < 8; ++r) {
        float x = a[r];
        hout[(size_t)(rbase + r) * CCH + c] = x;
        s += x; q += x * x;
      }
      s += __shfl_xor(s, 16, 32);   // fold lanes l and l+16 (same channel)
      q += __shfl_xor(q, 16, 32);
      if (lane < 16) atomicAdd(&stats[c], s);        // low half: sum
      else           atomicAdd(&stats[CCH + c], q);  // high half: sumsq
    }
  }
}

// ---------------------------------------------------------------------------
// Fold stats into per-channel affine: A = g*rsqrt(var+eps), B = b - mu*A
// ---------------------------------------------------------------------------
__global__ void bnfin_kernel(const float* __restrict__ stats,
                             const float* __restrict__ g,
                             const float* __restrict__ b,
                             float* __restrict__ AB) {
  int c = threadIdx.x;
  float mu  = stats[c] * (1.0f / NPTS);
  float var = stats[CCH + c] * (1.0f / NPTS) - mu * mu;
  float rs  = rsqrtf(var + EPSV);
  float A   = g[c] * rs;
  AB[c]       = A;
  AB[CCH + c] = b[c] - mu * A;
}

// ---------------------------------------------------------------------------
// acc (+)= sigmoid(h*A + B); mode 0: write, 1: accumulate, 2: acc + sig then
// multiply by features into out.
// ---------------------------------------------------------------------------
__global__ void apply_kernel(const float* __restrict__ h,
                             const float* __restrict__ AB,
                             const float* __restrict__ feat,
                             float* __restrict__ out, int mode) {
  size_t gid  = (size_t)blockIdx.x * blockDim.x + threadIdx.x;
  size_t base = gid * 4;
  int c = (int)(base & (CCH - 1));
  float4 hv = *(const float4*)(h + base);
  float4 sg;
  sg.x = 1.0f / (1.0f + __expf(-(hv.x * AB[c + 0] + AB[CCH + c + 0])));
  sg.y = 1.0f / (1.0f + __expf(-(hv.y * AB[c + 1] + AB[CCH + c + 1])));
  sg.z = 1.0f / (1.0f + __expf(-(hv.z * AB[c + 2] + AB[CCH + c + 2])));
  sg.w = 1.0f / (1.0f + __expf(-(hv.w * AB[c + 3] + AB[CCH + c + 3])));
  if (mode == 0) {
    *(float4*)(out + base) = sg;
  } else if (mode == 1) {
    float4 o = *(const float4*)(out + base);
    o.x += sg.x; o.y += sg.y; o.z += sg.z; o.w += sg.w;
    *(float4*)(out + base) = o;
  } else {
    float4 o = *(const float4*)(out + base);
    float4 f = *(const float4*)(feat + base);
    o.x = (o.x + sg.x) * f.x; o.y = (o.y + sg.y) * f.y;
    o.z = (o.z + sg.z) * f.z; o.w = (o.w + sg.w) * f.w;
    *(float4*)(out + base) = o;
  }
}

// ---------------------------------------------------------------------------
extern "C" void kernel_launch(void* const* d_in, const int* in_sizes, int n_in,
                              void* d_out, int out_size, void* d_ws, size_t ws_size,
                              hipStream_t stream) {
  const float* feat = (const float*)d_in[0];
  const float* W[3] = {(const float*)d_in[1], (const float*)d_in[2], (const float*)d_in[3]};
  const float* g[3] = {(const float*)d_in[4], (const float*)d_in[6], (const float*)d_in[8]};
  const float* b[3] = {(const float*)d_in[5], (const float*)d_in[7], (const float*)d_in[9]};
  const int* nbr[3] = {(const int*)d_in[10], (const int*)d_in[11], (const int*)d_in[12]};
  float* out = (float*)d_out;

  // Workspace layout
  const size_t wpackElems = (size_t)3 * 196608;          // f16
  _Float16* wpack = (_Float16*)d_ws;
  float* h     = (float*)((char*)d_ws + wpackElems * 2); // [N,256] f32
  float* stats = h + (size_t)NPTS * CCH;                 // [2,256]
  float* AB    = stats + 2 * CCH;                        // [2,256]

  // 1) pre-swizzle all 9 weight matrices into WMMA B-fragment order
  wpack_kernel<<<(int)(wpackElems / 256), 256, 0, stream>>>(W[0], W[1], W[2], wpack);

  // 2-4) per axis: gemm (+stats) -> bn fold -> sigmoid/accumulate
  for (int axis = 0; axis < 3; ++axis) {
    hipMemsetAsync(stats, 0, 2 * CCH * sizeof(float), stream);
    spconv_gemm<<<NPTS / 64, 256, 0, stream>>>(
        feat, wpack + (size_t)axis * 196608, nbr[axis], h, stats);
    bnfin_kernel<<<1, CCH, 0, stream>>>(stats, g[axis], b[axis], AB);
    apply_kernel<<<(NPTS * CCH / 4) / 256, 256, 0, stream>>>(h, AB, feat, out, axis);
  }
}
